// ChangePointKernelAttention_82343112999027
// MI455X (gfx1250) — matrix operand order repaired and checked by reading, not compiled, not run on hardware
//
#include <hip/hip_runtime.h>
#include <hip/hip_bf16.h>

typedef _Float16 f16;
typedef __attribute__((ext_vector_type(16))) _Float16 v16h;
typedef __attribute__((ext_vector_type(8)))  _Float16 v8h;
typedef __attribute__((ext_vector_type(8)))  float    v8f;

#define NB 2
#define NH 8
#define NS 2048
#define ND 64
#define NTOK (NB*NH*NS)

// scale = 2*sqrt(d) = 16 ; 1/scale = 0.0625 ; -2/sqrt(d) = -0.25
#define INV_SCALE 0.0625f
#define PI_F 3.14159265358979323846f

__device__ __forceinline__ v8f wmma16(v16h a, v16h b, v8f c) {
  return __builtin_amdgcn_wmma_f32_16x16x32_f16(false, a, false, b, (short)0, c, false, false);
}

// Build the 16-half A operand from two contiguous 8-half (16-byte) runs.
// Contiguous K order (matches B): lane half kh holds K = kh*16..+15;
// caller passes p = row + (chunk*32) + kh*16, second run is at p+8.
__device__ __forceinline__ v16h load_a16(const f16* p) {
  union { v16h v; v8h h[2]; } u;
  u.h[0] = *(const v8h*)(p);
  u.h[1] = *(const v8h*)(p + 8);
  return u.v;
}

// ---------------------------------------------------------------------------
// Precompute: f16 copies of q/k, normalized q/k, sigmoids, V^T, and scalars.
// One wave (32 lanes) per token; each lane handles 2 of the 64 dims.
// ---------------------------------------------------------------------------
__global__ __launch_bounds__(256)
void cpk_precompute(const float* __restrict__ q, const float* __restrict__ k,
                    const float* __restrict__ v,
                    f16* __restrict__ Qh,  f16* __restrict__ QNh, f16* __restrict__ SQh,
                    f16* __restrict__ Kh,  f16* __restrict__ KNh, f16* __restrict__ SKh,
                    f16* __restrict__ Vt,
                    float* __restrict__ q2a, float* __restrict__ k2a,
                    float* __restrict__ s1qa, float* __restrict__ s1ka)
{
  const int tok  = blockIdx.x * 8 + (threadIdx.x >> 5);
  const int lane = threadIdx.x & 31;
  if (tok >= NTOK) return;
  const int d0 = lane * 2;
  const size_t base = (size_t)tok * ND + d0;

  // ---- query side ----
  {
    float x0 = q[base], x1 = q[base + 1];
    float ss = x0 * x0 + x1 * x1;
    #pragma unroll
    for (int m = 1; m < 32; m <<= 1) ss += __shfl_xor(ss, m);
    float inorm = rsqrtf(ss);
    float n0 = x0 * inorm, n1 = x1 * inorm;
    float g0 = 1.0f / (1.0f + __expf(-n0));
    float g1 = 1.0f / (1.0f + __expf(-n1));
    float sg = g0 + g1;
    #pragma unroll
    for (int m = 1; m < 32; m <<= 1) sg += __shfl_xor(sg, m);
    Qh[base]  = (f16)x0; Qh[base + 1]  = (f16)x1;
    QNh[base] = (f16)n0; QNh[base + 1] = (f16)n1;
    SQh[base] = (f16)g0; SQh[base + 1] = (f16)g1;
    if (lane == 0) { q2a[tok] = ss; s1qa[tok] = sg; }
  }
  // ---- key side ----
  {
    float x0 = k[base], x1 = k[base + 1];
    float ss = x0 * x0 + x1 * x1;
    #pragma unroll
    for (int m = 1; m < 32; m <<= 1) ss += __shfl_xor(ss, m);
    float inorm = rsqrtf(ss);
    float n0 = x0 * inorm, n1 = x1 * inorm;
    float g0 = 1.0f / (1.0f + __expf(-n0));
    float g1 = 1.0f / (1.0f + __expf(-n1));
    float sg = g0 + g1;
    #pragma unroll
    for (int m = 1; m < 32; m <<= 1) sg += __shfl_xor(sg, m);
    Kh[base]  = (f16)x0; Kh[base + 1]  = (f16)x1;
    KNh[base] = (f16)n0; KNh[base + 1] = (f16)n1;
    SKh[base] = (f16)g0; SKh[base + 1] = (f16)g1;
    if (lane == 0) { k2a[tok] = ss; s1ka[tok] = sg; }
  }
  // ---- V transpose to f16:  Vt[bh][d][s] = V[bh][s][d] ----
  {
    const int bh = tok / NS;
    const int s  = tok & (NS - 1);
    f16* vt = Vt + (size_t)bh * ND * NS;
    vt[(size_t)(d0    ) * NS + s] = (f16)v[base];
    vt[(size_t)(d0 + 1) * NS + s] = (f16)v[base + 1];
  }
}

// ---------------------------------------------------------------------------
// Fused attention: per wave one 16-row q tile, single pass over k (step 32).
// Score GEMMs + elementwise + S@V and sum(S^2) accumulated on the fly;
// final O /= max(sqrt(ss), eps).
// ---------------------------------------------------------------------------
__global__ __launch_bounds__(256)
void cpk_attn(const f16* __restrict__ Qh,  const f16* __restrict__ QNh, const f16* __restrict__ SQh,
              const f16* __restrict__ Kh,  const f16* __restrict__ KNh, const f16* __restrict__ SKh,
              const f16* __restrict__ Vt,
              const float* __restrict__ q2a, const float* __restrict__ k2a,
              const float* __restrict__ s1qa, const float* __restrict__ s1ka,
              float* __restrict__ out)
{
  __shared__ f16 sS[8][16][32];   // per-wave S tile staging (f32 C-layout -> f16 A-layout)

  const int lane = threadIdx.x & 31;
  const int wid  = threadIdx.x >> 5;
  const int bh   = blockIdx.y;
  const int q0   = blockIdx.x * 128 + wid * 16;
  const int n    = lane & 15;     // C/B-layout column, A-layout row
  const int kh   = lane >> 4;     // lane half

  const size_t tb = (size_t)bh * NS;
  const f16* Qh_b  = Qh  + tb * ND;
  const f16* QNh_b = QNh + tb * ND;  (void)QNh_b;
  const f16* SQh_b = SQh + tb * ND;
  const f16* Kh_b  = Kh  + tb * ND;
  const f16* KNh_b = KNh + tb * ND;
  const f16* SKh_b = SKh + tb * ND;
  const f16* Vt_b  = Vt  + (size_t)bh * ND * NS;
  const float* q2_b  = q2a  + tb;
  const float* k2_b  = k2a  + tb;
  const float* s1q_b = s1qa + tb;
  const float* s1k_b = s1ka + tb;

  // Loop-invariant A operands (q-tile rows). Transposed-SE trick: the SE GEMM's
  // A side is KEYS at q rows; its B side is QUERY at k cols (QK[k,q]).
  const f16* rG = Kh_b          + (size_t)(q0 + n) * ND;   // keys  @ q rows
  const f16* rC = (QNh + tb*ND) + (size_t)(q0 + n) * ND;   // qn    @ q rows
  const f16* rP = SQh_b         + (size_t)(q0 + n) * ND;   // s(qn) @ q rows
  v16h AG0 = load_a16(rG + kh * 16),      AG1 = load_a16(rG + 32 + kh * 16);
  v16h AC0 = load_a16(rC + kh * 16),      AC1 = load_a16(rC + 32 + kh * 16);
  v16h AP0 = load_a16(rP + kh * 16),      AP1 = load_a16(rP + 32 + kh * 16);

  // q-row scalars for C-layout rows m = r + 8*kh
  float q2q[8], k2q[8], s1q[8];
  #pragma unroll
  for (int r = 0; r < 8; ++r) {
    const int qi = q0 + r + 8 * kh;
    q2q[r] = q2_b[qi]; k2q[r] = k2_b[qi]; s1q[r] = s1q_b[qi];
  }

  v8f o0 = {0,0,0,0,0,0,0,0}, o1 = {0,0,0,0,0,0,0,0};
  v8f o2 = {0,0,0,0,0,0,0,0}, o3 = {0,0,0,0,0,0,0,0};
  float ssacc[8] = {0,0,0,0,0,0,0,0};

  for (int kk = 0; kk < NS; kk += 32) {
    #pragma unroll
    for (int sub = 0; sub < 2; ++sub) {
      const int kt = kk + sub * 16 + n;           // B-operand column token
      const f16* bG = Qh_b  + (size_t)kt * ND;    // query @ k cols (transposed SE)
      const f16* bC = KNh_b + (size_t)kt * ND;
      const f16* bP = SKh_b + (size_t)kt * ND;
      // B operand in this code's contiguous K order (same as A): lane half kh holds K = kh*16..+15 (contiguous in d)
      v16h BG0 = *(const v16h*)(bG + kh * 16), BG1 = *(const v16h*)(bG + 32 + kh * 16);
      v16h BC0 = *(const v16h*)(bC + kh * 16), BC1 = *(const v16h*)(bC + 32 + kh * 16);
      v16h BP0 = *(const v16h*)(bP + kh * 16), BP1 = *(const v16h*)(bP + 32 + kh * 16);
      const float q2k = q2_b[kt], k2k = k2_b[kt], s1k = s1k_b[kt];

      v8f aG = {0,0,0,0,0,0,0,0}, aC = {0,0,0,0,0,0,0,0}, aP = {0,0,0,0,0,0,0,0};
      aG = wmma16(AG0, BG0, aG);  aG = wmma16(AG1, BG1, aG);
      aC = wmma16(AC0, BC0, aC);  aC = wmma16(AC1, BC1, aC);
      aP = wmma16(AP0, BP0, aP);  aP = wmma16(AP1, BP1, aP);

      #pragma unroll
      for (int r = 0; r < 8; ++r) {
        // SE (transposed index): d2 = q2[k_col] + k2[q_row] - 2*keys[q]·query[k]
        float d2  = q2k + k2q[r] - 2.0f * aG[r];
        float se  = -sqrtf(fmaxf(d2, 0.0f)) * INV_SCALE;
        // periodic kernel
        float cs  = aC[r];
        float pre = PI_F * sqrtf(fmaxf(2.0f - 2.0f * cs, 0.0f));
        float sn  = __sinf(pre);
        float per = -0.25f * sn * sn;               // -2*sin^2/sqrt(64)
        // gates
        float gp  = aP[r];
        float gn  = 64.0f - s1q[r] - s1k + gp;
        // preattn + magnitude term (non-transposed indices)
        float sv  = gp * __expf(se) + gn * __expf(per) + (q2q[r] + k2k) * INV_SCALE;
        ssacc[r] += sv * sv;
        sS[wid][r + 8 * kh][sub * 16 + n] = (f16)sv;
      }
    }

    // PV GEMM: A = S tile (16q x 32k) from LDS (A-layout), B = V^T chunks.
    const f16* srow = &sS[wid][n][0];
    union { v16h v; v8h h[2]; } ua;
    ua.h[0] = *(const v8h*)(srow + kh * 16);
    ua.h[1] = *(const v8h*)(srow + kh * 16 + 8);
    const v16h AS = ua.v;

    v16h BV0 = *(const v16h*)(Vt_b + (size_t)( 0 + n) * NS + kk + kh * 16);
    v16h BV1 = *(const v16h*)(Vt_b + (size_t)(16 + n) * NS + kk + kh * 16);
    v16h BV2 = *(const v16h*)(Vt_b + (size_t)(32 + n) * NS + kk + kh * 16);
    v16h BV3 = *(const v16h*)(Vt_b + (size_t)(48 + n) * NS + kk + kh * 16);
    o0 = wmma16(AS, BV0, o0);
    o1 = wmma16(AS, BV1, o1);
    o2 = wmma16(AS, BV2, o2);
    o3 = wmma16(AS, BV3, o3);
  }

  // Row L2 norm: reduce ss over the 16 lanes of this half (masks < 16 keep bit4)
  #pragma unroll
  for (int r = 0; r < 8; ++r) {
    float s = ssacc[r];
    s += __shfl_xor(s, 1); s += __shfl_xor(s, 2);
    s += __shfl_xor(s, 4); s += __shfl_xor(s, 8);
    ssacc[r] = s;
  }

  float* out_b = out + tb * ND;
  #pragma unroll
  for (int r = 0; r < 8; ++r) {
    const float rn = 1.0f / fmaxf(sqrtf(ssacc[r]), 1e-12f);
    float* orow = out_b + (size_t)(q0 + r + 8 * kh) * ND;
    orow[ 0 + n] = o0[r] * rn;
    orow[16 + n] = o1[r] * rn;
    orow[32 + n] = o2[r] * rn;
    orow[48 + n] = o3[r] * rn;
  }
}

// ---------------------------------------------------------------------------
extern "C" void kernel_launch(void* const* d_in, const int* in_sizes, int n_in,
                              void* d_out, int out_size, void* d_ws, size_t ws_size,
                              hipStream_t stream) {
  const float* q = (const float*)d_in[0];
  const float* k = (const float*)d_in[1];
  const float* v = (const float*)d_in[2];
  float* out = (float*)d_out;

  const size_t N64 = (size_t)NTOK * ND;
  f16* Qh   = (f16*)d_ws;
  f16* QNh  = Qh  + N64;
  f16* SQh  = QNh + N64;
  f16* Kh   = SQh + N64;
  f16* KNh  = Kh  + N64;
  f16* SKh  = KNh + N64;
  f16* Vt   = SKh + N64;
  float* q2a  = (float*)(Vt + N64);
  float* k2a  = q2a  + NTOK;
  float* s1qa = k2a  + NTOK;
  float* s1ka = s1qa + NTOK;

  cpk_precompute<<<NTOK / 8, 256, 0, stream>>>(q, k, v, Qh, QNh, SQh, Kh, KNh, SKh,
                                               Vt, q2a, k2a, s1qa, s1ka);
  dim3 grid(NS / 128, NB * NH);
  cpk_attn<<<grid, 256, 0, stream>>>(Qh, QNh, SQh, Kh, KNh, SKh, Vt,
                                     q2a, k2a, s1qa, s1ka, out);
}
